// ResGatedGraphConvRefactored_21534966022785
// MI455X (gfx1250) — matrix operand order repaired and checked by reading, not compiled, not run on hardware
//
#include <hip/hip_runtime.h>
#include <cstdint>
#include <cstddef>

// ---------------- types for WMMA (wave32, gfx1250) ----------------
typedef __attribute__((ext_vector_type(16))) __bf16 v16bf;
typedef __attribute__((ext_vector_type(8)))  __bf16 v8bf;
typedef __attribute__((ext_vector_type(8)))  float  v8f;

#define N_NODES  100000
#define N_EDGES  1000000
#define F_INN    128
#define F_HID    64
#define N_GRAPHS 128
#define BN_EPS   1e-5f
#define THREADS  256

__device__ __forceinline__ float sigmoidf(float x) { return 1.0f / (1.0f + __expf(-x)); }

// ---------------- utility kernels ----------------
__global__ void k_fill(float* __restrict__ p, int n, float v) {
    int i = blockIdx.x * blockDim.x + threadIdx.x;
    if (i < n) p[i] = v;
}

__global__ void k_cvt_bf16(const float* __restrict__ src, __bf16* __restrict__ dst, int n) {
    int i = blockIdx.x * blockDim.x + threadIdx.x;
    if (i < n) dst[i] = (__bf16)src[i];
}

// W is (K x Nout) row-major f32; dst is col-major bf16: dst[n*K + k]
__global__ void k_pack_w(const float* __restrict__ W, __bf16* __restrict__ dst, int K, int Nout) {
    int i = blockIdx.x * blockDim.x + threadIdx.x;
    if (i < K * Nout) {
        int n = i / K, k = i - n * K;
        dst[i] = (__bf16)W[(size_t)k * Nout + n];
    }
}

__global__ void k_concat4(const float* __restrict__ a, const float* __restrict__ b,
                          const float* __restrict__ c, const float* __restrict__ d,
                          float* __restrict__ out, int L) {
    int i = blockIdx.x * blockDim.x + threadIdx.x;
    if (i < 4 * L) {
        int which = i / L, j = i - which * L;
        const float* s = (which == 0) ? a : (which == 1) ? b : (which == 2) ? c : d;
        out[i] = s[j];
    }
}

// ---------------- WMMA GEMM ----------------
// C[M,Nout] = A[M,K](bf16,row-major) * B[K,Nout](bf16, packed col-major Bc[n*K+k]) + bias[n]
// Each wave computes a 16x64 strip: one A fragment feeds FOUR v_wmma_f32_16x16x32_bf16,
// software-pipelined so the next K-step's 10 b128 loads are in flight during this step's WMMAs.
// 256-thread blocks = 8 waves (fills a WGP 2-deep).
//
// Fragment layouts (wave32, ISA 7.12.2):
//  A (16x32): half = lane>>4, row m = lane&15;
//     elems e -> K = kk + 16*half + e (contiguous, matches B)
//  B (32x16): col n = lane&15; elems e -> K = kk + 16*half + e (contiguous in col-major pack)
//  C/D: VGPR r -> row = 16*mt + r + 8*half, col = lane&15 within tile
__device__ __forceinline__ v16bf ld_frag16(const __bf16* p0, const __bf16* p1) {
    v8bf lo = *(const v8bf*)p0;
    v8bf hi = *(const v8bf*)p1;
    return __builtin_shufflevector(lo, hi, 0, 1, 2, 3, 4, 5, 6, 7, 8, 9, 10, 11, 12, 13, 14, 15);
}

__global__ void k_wmma_gemm4(const __bf16* __restrict__ A, const __bf16* __restrict__ Bc,
                             const float* __restrict__ bias, float* __restrict__ C,
                             int M, int K, int Nout, int totalWaveTiles) {
    int wave = threadIdx.x >> 5;
    int lane = threadIdx.x & 31;
    int wt = blockIdx.x * 8 + wave;            // wave-uniform branch -> EXEC all-ones for WMMA
    if (wt >= totalWaveTiles) return;
    int groupsN = Nout >> 6;                   // 64 columns per wave
    int mt = wt / groupsN, ng = wt - mt * groupsN;
    int half = lane >> 4, l16 = lane & 15;

    const __bf16* arow = A + (size_t)(mt * 16 + l16) * K;
    const __bf16* bcol0 = Bc + (size_t)(ng * 64 + l16) * K;   // tile t adds 16*K

    v8f acc0, acc1, acc2, acc3;
    {
        float b0 = bias[ng * 64 + 0  + l16];
        float b1 = bias[ng * 64 + 16 + l16];
        float b2 = bias[ng * 64 + 32 + l16];
        float b3 = bias[ng * 64 + 48 + l16];
#pragma unroll
        for (int r = 0; r < 8; ++r) { acc0[r] = b0; acc1[r] = b1; acc2[r] = b2; acc3[r] = b3; }
    }

    // --- prologue: load K-step 0 fragments ---
    v16bf aC, b0C, b1C, b2C, b3C;
    aC  = ld_frag16(arow + 16 * half,               arow + 16 * half + 8);
    b0C = ld_frag16(bcol0 + 0 * 16 * K + 16 * half, bcol0 + 0 * 16 * K + 16 * half + 8);
    b1C = ld_frag16(bcol0 + 1 * 16 * K + 16 * half, bcol0 + 1 * 16 * K + 16 * half + 8);
    b2C = ld_frag16(bcol0 + 2 * 16 * K + 16 * half, bcol0 + 2 * 16 * K + 16 * half + 8);
    b3C = ld_frag16(bcol0 + 3 * 16 * K + 16 * half, bcol0 + 3 * 16 * K + 16 * half + 8);

    int kk = 0;
    for (; kk + 32 < K; kk += 32) {
        int kn = kk + 32;
        // WGP-scope prefetch of the A stream (locality 3 -> nearest cache level)
        __builtin_prefetch(arow + kn + 32, 0, 3);
        // issue next-step loads BEFORE this step's WMMAs (double buffer)
        v16bf aN  = ld_frag16(arow + kn + 16 * half,               arow + kn + 16 * half + 8);
        v16bf b0N = ld_frag16(bcol0 + 0 * 16 * K + kn + 16 * half, bcol0 + 0 * 16 * K + kn + 16 * half + 8);
        v16bf b1N = ld_frag16(bcol0 + 1 * 16 * K + kn + 16 * half, bcol0 + 1 * 16 * K + kn + 16 * half + 8);
        v16bf b2N = ld_frag16(bcol0 + 2 * 16 * K + kn + 16 * half, bcol0 + 2 * 16 * K + kn + 16 * half + 8);
        v16bf b3N = ld_frag16(bcol0 + 3 * 16 * K + kn + 16 * half, bcol0 + 3 * 16 * K + kn + 16 * half + 8);

        acc0 = __builtin_amdgcn_wmma_f32_16x16x32_bf16(false, aC, false, b0C, (short)0, acc0, false, false);
        acc1 = __builtin_amdgcn_wmma_f32_16x16x32_bf16(false, aC, false, b1C, (short)0, acc1, false, false);
        acc2 = __builtin_amdgcn_wmma_f32_16x16x32_bf16(false, aC, false, b2C, (short)0, acc2, false, false);
        acc3 = __builtin_amdgcn_wmma_f32_16x16x32_bf16(false, aC, false, b3C, (short)0, acc3, false, false);

        aC = aN; b0C = b0N; b1C = b1N; b2C = b2N; b3C = b3N;
    }
    // --- epilogue: last K-step ---
    acc0 = __builtin_amdgcn_wmma_f32_16x16x32_bf16(false, aC, false, b0C, (short)0, acc0, false, false);
    acc1 = __builtin_amdgcn_wmma_f32_16x16x32_bf16(false, aC, false, b1C, (short)0, acc1, false, false);
    acc2 = __builtin_amdgcn_wmma_f32_16x16x32_bf16(false, aC, false, b2C, (short)0, acc2, false, false);
    acc3 = __builtin_amdgcn_wmma_f32_16x16x32_bf16(false, aC, false, b3C, (short)0, acc3, false, false);

    float* cp = C + (size_t)(mt * 16 + 8 * half) * Nout + ng * 64 + l16;
#pragma unroll
    for (int r = 0; r < 8; ++r) {
        cp[(size_t)r * Nout + 0]  = acc0[r];
        cp[(size_t)r * Nout + 16] = acc1[r];
        cp[(size_t)r * Nout + 32] = acc2[r];
        cp[(size_t)r * Nout + 48] = acc3[r];
    }
}

// ---------------- graph-conv kernels ----------------
// C layout per node row (256): [k 0..63 | q 64..127 | v 128..191 | s 192..255]
__global__ void k_init_agg(const float* __restrict__ C, float* __restrict__ agg, int N) {
    int i = blockIdx.x * blockDim.x + threadIdx.x;
    if (i < N * F_HID) {
        int nd = i >> 6, f = i & 63;
        agg[i] = C[(size_t)nd * 256 + 192 + f];
    }
}

__global__ void k_edge_msg(const int* __restrict__ ei, const float* __restrict__ C,
                           float* __restrict__ agg, int E) {
    int i = blockIdx.x * blockDim.x + threadIdx.x;  // E * 16 work items (float4 each)
    if (i >= E * 16) return;
    int e = i >> 4, f = (i & 15) << 2;
    int s = ei[e], d = ei[E + e];
    float4 kv = *(const float4*)(C + (size_t)d * 256 + 0   + f);
    float4 qv = *(const float4*)(C + (size_t)s * 256 + 64  + f);
    float4 vv = *(const float4*)(C + (size_t)s * 256 + 128 + f);
    float4 mg;
    mg.x = sigmoidf(kv.x + qv.x) * vv.x;
    mg.y = sigmoidf(kv.y + qv.y) * vv.y;
    mg.z = sigmoidf(kv.z + qv.z) * vv.z;
    mg.w = sigmoidf(kv.w + qv.w) * vv.w;
    float* ap = agg + (size_t)d * F_HID + f;
    atomicAdd(ap + 0, mg.x);
    atomicAdd(ap + 1, mg.y);
    atomicAdd(ap + 2, mg.z);
    atomicAdd(ap + 3, mg.w);
}

// ---------------- batchnorm over N nodes (stats of sigmoid(agg)) ----------------
// LDS-staged partial sums: ds_add_f32 per thread, then 128 global atomics per block.
__global__ void k_bn_reduce(const float* __restrict__ agg, float* __restrict__ stats, int N) {
    __shared__ float sh[128];
    if (threadIdx.x < 128) sh[threadIdx.x] = 0.f;
    __syncthreads();
    int f = threadIdx.x & 63;
    int row0 = blockIdx.x * (blockDim.x >> 6) + (threadIdx.x >> 6);
    int rstride = gridDim.x * (blockDim.x >> 6);
    float s = 0.f, s2 = 0.f;
    for (int nd = row0; nd < N; nd += rstride) {
        float z = sigmoidf(agg[(size_t)nd * 64 + f]);
        s += z; s2 += z * z;
    }
    atomicAdd(&sh[f], s);        // LDS atomic (ds_add_f32)
    atomicAdd(&sh[64 + f], s2);
    __syncthreads();
    if (threadIdx.x < 128) atomicAdd(&stats[threadIdx.x], sh[threadIdx.x]);
}

__global__ void k_bn_apply(const float* __restrict__ agg, const float* __restrict__ stats,
                           const float* __restrict__ gain, const float* __restrict__ beta,
                           float* __restrict__ h, __bf16* __restrict__ hbf, int N) {
    int i = blockIdx.x * blockDim.x + threadIdx.x;
    if (i >= N * 64) return;
    int f = i & 63;
    float inv_n = 1.0f / (float)N;
    float mean = stats[f] * inv_n;
    float var  = stats[64 + f] * inv_n - mean * mean;
    float z = sigmoidf(agg[i]);
    float y = (z - mean) * rsqrtf(var + BN_EPS) * gain[f] + beta[f];
    h[i] = y;
    if (hbf) hbf[i] = (__bf16)y;
}

// ---------------- pooling ----------------
__global__ void k_count(const int* __restrict__ batch, float* __restrict__ counts, int N) {
    int i = blockIdx.x * blockDim.x + threadIdx.x;
    if (i < N) atomicAdd(&counts[batch[i]], 1.0f);
}

__global__ void k_pool_sum(const float* __restrict__ h, const int* __restrict__ batch,
                           float* __restrict__ sum, int N) {
    int i = blockIdx.x * blockDim.x + threadIdx.x;
    if (i >= N * 64) return;
    int nd = i >> 6, f = i & 63;
    atomicAdd(&sum[(size_t)batch[nd] * 64 + f], h[i]);
}

// out0[G,256] = [mean h0 | mean h1 | sum h0 | sum h1]
__global__ void k_build_out0(const float* __restrict__ s0, const float* __restrict__ s1,
                             const float* __restrict__ counts,
                             float* __restrict__ out0, __bf16* __restrict__ out0bf) {
    int i = blockIdx.x * blockDim.x + threadIdx.x;
    if (i >= N_GRAPHS * 256) return;
    int g = i >> 8, c = i & 255;
    float cnt = counts[g]; cnt = (cnt < 1.f) ? 1.f : cnt;
    float v;
    if (c < 64)       v = s0[g * 64 + c] / cnt;
    else if (c < 128) v = s1[g * 64 + (c - 64)] / cnt;
    else if (c < 192) v = s0[g * 64 + (c - 128)];
    else              v = s1[g * 64 + (c - 192)];
    out0[i] = v;
    out0bf[i] = (__bf16)v;
}

// BN over G=128 rows, F features; one block, one feature per thread.
__global__ void k_bn_small(const float* __restrict__ raw, const float* __restrict__ gain,
                           const float* __restrict__ beta, float* __restrict__ outf,
                           __bf16* __restrict__ outbf, int rows, int F) {
    int f = threadIdx.x;
    if (f >= F) return;
    float s = 0.f, s2 = 0.f;
    for (int r = 0; r < rows; ++r) {
        float z = sigmoidf(raw[(size_t)r * F + f]);
        s += z; s2 += z * z;
    }
    float mean = s / rows, var = s2 / rows - mean * mean;
    float sc = rsqrtf(var + BN_EPS) * gain[f];
    for (int r = 0; r < rows; ++r) {
        float z = sigmoidf(raw[(size_t)r * F + f]);
        float y = (z - mean) * sc + beta[f];
        outf[(size_t)r * F + f] = y;
        outbf[(size_t)r * F + f] = (__bf16)y;
    }
}

__global__ void k_classifier(const float* __restrict__ t, const float* __restrict__ Wc,
                             const float* __restrict__ bc, float* __restrict__ out) {
    int i = blockIdx.x * blockDim.x + threadIdx.x;  // G*2
    if (i >= N_GRAPHS * 2) return;
    int g = i >> 1, o = i & 1;
    float acc = bc[o];
    for (int k2 = 0; k2 < 64; ++k2) acc += t[g * 64 + k2] * Wc[k2 * 2 + o];
    out[i] = acc;
}

// ---------------- launch ----------------
static inline int blocks_for(long long n) { return (int)((n + THREADS - 1) / THREADS); }

extern "C" void kernel_launch(void* const* d_in, const int* in_sizes, int n_in,
                              void* d_out, int out_size, void* d_ws, size_t ws_size,
                              hipStream_t stream) {
    const float* x    = (const float*)d_in[0];
    const int*   ei   = (const int*)d_in[1];
    const int*   batch= (const int*)d_in[2];
    const float* Wk0 = (const float*)d_in[3],  *bk0 = (const float*)d_in[4];
    const float* Wq0 = (const float*)d_in[5],  *bq0 = (const float*)d_in[6];
    const float* Wv0 = (const float*)d_in[7],  *bv0 = (const float*)d_in[8];
    const float* Ws0 = (const float*)d_in[9],  *cb0 = (const float*)d_in[10];
    const float* g0  = (const float*)d_in[11], *be0 = (const float*)d_in[12];
    const float* Wk1 = (const float*)d_in[13], *bk1 = (const float*)d_in[14];
    const float* Wq1 = (const float*)d_in[15], *bq1 = (const float*)d_in[16];
    const float* Wv1 = (const float*)d_in[17], *bv1 = (const float*)d_in[18];
    const float* Ws1 = (const float*)d_in[19], *cb1 = (const float*)d_in[20];
    const float* g1  = (const float*)d_in[21], *be1 = (const float*)d_in[22];
    const float* Wh0 = (const float*)d_in[23], *bh0 = (const float*)d_in[24];
    const float* gh0 = (const float*)d_in[25], *beh0= (const float*)d_in[26];
    const float* Wh1 = (const float*)d_in[27], *bh1 = (const float*)d_in[28];
    const float* gh1 = (const float*)d_in[29], *beh1= (const float*)d_in[30];
    const float* Wc  = (const float*)d_in[31], *bc  = (const float*)d_in[32];
    float* out = (float*)d_out;

    const int N = N_NODES, E = N_EDGES;

    // workspace carve-up (256B aligned slices)
    char* w = (char*)d_ws;
    auto carve = [&](size_t bytes) -> char* {
        char* p = w;
        w += (bytes + 255) & ~(size_t)255;
        return p;
    };
    float*  C0     = (float*)carve((size_t)N * 256 * sizeof(float));   // kqvs (reused for layer1)
    __bf16* Xbf    = (__bf16*)carve((size_t)N * F_INN * sizeof(__bf16));
    float*  agg    = (float*)carve((size_t)N * F_HID * sizeof(float));
    float*  h0f    = (float*)carve((size_t)N * F_HID * sizeof(float));
    float*  h1f    = (float*)carve((size_t)N * F_HID * sizeof(float));
    __bf16* h0bf   = (__bf16*)Xbf;                                     // Xbf dead after layer0 GEMM
    __bf16* W0pack = (__bf16*)carve((size_t)256 * 128 * sizeof(__bf16));
    __bf16* W1pack = (__bf16*)carve((size_t)256 * 64  * sizeof(__bf16));
    __bf16* Wh0pk  = (__bf16*)carve((size_t)128 * 256 * sizeof(__bf16));
    __bf16* Wh1pk  = (__bf16*)carve((size_t)64  * 128 * sizeof(__bf16));
    float*  bias0  = (float*)carve(256 * sizeof(float));
    float*  bias1  = (float*)carve(256 * sizeof(float));
    float*  stats  = (float*)carve(128 * sizeof(float));
    float*  counts = (float*)carve(N_GRAPHS * sizeof(float));
    float*  sum0   = (float*)carve((size_t)N_GRAPHS * 64 * sizeof(float));
    float*  sum1   = (float*)carve((size_t)N_GRAPHS * 64 * sizeof(float));
    float*  out0f  = (float*)carve((size_t)N_GRAPHS * 256 * sizeof(float));
    __bf16* out0bf = (__bf16*)carve((size_t)N_GRAPHS * 256 * sizeof(__bf16));
    float*  t0raw  = (float*)carve((size_t)N_GRAPHS * 128 * sizeof(float));
    float*  t0f    = (float*)carve((size_t)N_GRAPHS * 128 * sizeof(float));
    __bf16* t0bf   = (__bf16*)carve((size_t)N_GRAPHS * 128 * sizeof(__bf16));
    float*  t1raw  = (float*)carve((size_t)N_GRAPHS * 64 * sizeof(float));
    float*  t1f    = (float*)carve((size_t)N_GRAPHS * 64 * sizeof(float));
    __bf16* t1bf   = (__bf16*)carve((size_t)N_GRAPHS * 64 * sizeof(__bf16));
    (void)ws_size; (void)in_sizes; (void)n_in; (void)out_size;

    // ---------- layer 0 ----------
    k_cvt_bf16<<<blocks_for((long long)N * F_INN), THREADS, 0, stream>>>(x, Xbf, N * F_INN);
    k_pack_w<<<blocks_for(128 * 64), THREADS, 0, stream>>>(Wk0, W0pack + (size_t)0   * 128, 128, 64);
    k_pack_w<<<blocks_for(128 * 64), THREADS, 0, stream>>>(Wq0, W0pack + (size_t)64  * 128, 128, 64);
    k_pack_w<<<blocks_for(128 * 64), THREADS, 0, stream>>>(Wv0, W0pack + (size_t)128 * 128, 128, 64);
    k_pack_w<<<blocks_for(128 * 64), THREADS, 0, stream>>>(Ws0, W0pack + (size_t)192 * 128, 128, 64);
    k_concat4<<<1, THREADS, 0, stream>>>(bk0, bq0, bv0, cb0, bias0, 64);

    int wt0 = (N / 16) * (256 / 64);   // 16-row x 64-col strips
    k_wmma_gemm4<<<(wt0 + 7) / 8, 256, 0, stream>>>(Xbf, W0pack, bias0, C0, N, 128, 256, wt0);

    k_init_agg<<<blocks_for((long long)N * 64), THREADS, 0, stream>>>(C0, agg, N);
    k_edge_msg<<<blocks_for((long long)E * 16), THREADS, 0, stream>>>(ei, C0, agg, E);

    k_fill<<<1, 128, 0, stream>>>(stats, 128, 0.f);
    k_bn_reduce<<<1024, THREADS, 0, stream>>>(agg, stats, N);
    k_bn_apply<<<blocks_for((long long)N * 64), THREADS, 0, stream>>>(agg, stats, g0, be0, h0f, h0bf, N);

    // ---------- layer 1 ----------
    k_pack_w<<<blocks_for(64 * 64), THREADS, 0, stream>>>(Wk1, W1pack + (size_t)0   * 64, 64, 64);
    k_pack_w<<<blocks_for(64 * 64), THREADS, 0, stream>>>(Wq1, W1pack + (size_t)64  * 64, 64, 64);
    k_pack_w<<<blocks_for(64 * 64), THREADS, 0, stream>>>(Wv1, W1pack + (size_t)128 * 64, 64, 64);
    k_pack_w<<<blocks_for(64 * 64), THREADS, 0, stream>>>(Ws1, W1pack + (size_t)192 * 64, 64, 64);
    k_concat4<<<1, THREADS, 0, stream>>>(bk1, bq1, bv1, cb1, bias1, 64);

    int wt1 = (N / 16) * (256 / 64);
    k_wmma_gemm4<<<(wt1 + 7) / 8, 256, 0, stream>>>(h0bf, W1pack, bias1, C0, N, 64, 256, wt1);

    k_init_agg<<<blocks_for((long long)N * 64), THREADS, 0, stream>>>(C0, agg, N);
    k_edge_msg<<<blocks_for((long long)E * 16), THREADS, 0, stream>>>(ei, C0, agg, E);

    k_fill<<<1, 128, 0, stream>>>(stats, 128, 0.f);
    k_bn_reduce<<<1024, THREADS, 0, stream>>>(agg, stats, N);
    k_bn_apply<<<blocks_for((long long)N * 64), THREADS, 0, stream>>>(agg, stats, g1, be1, h1f, (__bf16*)nullptr, N);

    // ---------- pooling ----------
    k_fill<<<1, 128, 0, stream>>>(counts, N_GRAPHS, 0.f);
    k_fill<<<blocks_for(N_GRAPHS * 64), THREADS, 0, stream>>>(sum0, N_GRAPHS * 64, 0.f);
    k_fill<<<blocks_for(N_GRAPHS * 64), THREADS, 0, stream>>>(sum1, N_GRAPHS * 64, 0.f);
    k_count<<<blocks_for(N), THREADS, 0, stream>>>(batch, counts, N);
    k_pool_sum<<<blocks_for((long long)N * 64), THREADS, 0, stream>>>(h0f, batch, sum0, N);
    k_pool_sum<<<blocks_for((long long)N * 64), THREADS, 0, stream>>>(h1f, batch, sum1, N);
    k_build_out0<<<blocks_for(N_GRAPHS * 256), THREADS, 0, stream>>>(sum0, sum1, counts, out0f, out0bf);

    // ---------- hidden MLP (WMMA) ----------
    k_pack_w<<<blocks_for(256 * 128), THREADS, 0, stream>>>(Wh0, Wh0pk, 256, 128);
    int wtH0 = (N_GRAPHS / 16) * (128 / 64);
    k_wmma_gemm4<<<(wtH0 + 7) / 8, 256, 0, stream>>>(out0bf, Wh0pk, bh0, t0raw, N_GRAPHS, 256, 128, wtH0);
    k_bn_small<<<1, 128, 0, stream>>>(t0raw, gh0, beh0, t0f, t0bf, N_GRAPHS, 128);

    k_pack_w<<<blocks_for(128 * 64), THREADS, 0, stream>>>(Wh1, Wh1pk, 128, 64);
    int wtH1 = (N_GRAPHS / 16) * (64 / 64);
    k_wmma_gemm4<<<(wtH1 + 7) / 8, 256, 0, stream>>>(t0bf, Wh1pk, bh1, t1raw, N_GRAPHS, 128, 64, wtH1);
    k_bn_small<<<1, 64, 0, stream>>>(t1raw, gh1, beh1, t1f, t1bf, N_GRAPHS, 64);

    // ---------- classifier ----------
    k_classifier<<<1, THREADS, 0, stream>>>(t1f, Wc, bc, out);
}
